// DecoderModule_29643864277330
// MI455X (gfx1250) — matrix operand order repaired and checked by reading, not compiled, not run on hardware
//
#include <hip/hip_runtime.h>
#include <hip/hip_bf16.h>

typedef __bf16 bf16;
typedef __attribute__((ext_vector_type(16))) __bf16 v16bf;
typedef __attribute__((ext_vector_type(8)))  __bf16 v8bf;
typedef __attribute__((ext_vector_type(8)))  float  v8f;
typedef __attribute__((ext_vector_type(4)))  int    v4i;

constexpr int T_ = 2048;
constexpr int E_ = 1024;
constexpr int H_ = 16;
constexpr int D_ = 64;
constexpr int F_ = 4096;

// ---------------------------------------------------------------------------
// CDNA5 async global->LDS staging (ASYNCcnt path), with sync fallback
// ---------------------------------------------------------------------------
#if defined(__has_builtin)
#if __has_builtin(__builtin_amdgcn_global_load_async_to_lds_b128) && \
    __has_builtin(__builtin_amdgcn_s_wait_asynccnt)
#define HAVE_ASYNC 1
#endif
#endif
#ifndef HAVE_ASYNC
#define HAVE_ASYNC 0
#endif

__device__ inline void stage16(const bf16* g, bf16* l) {
#if HAVE_ASYNC
  __builtin_amdgcn_global_load_async_to_lds_b128((v4i*)(g), (v4i*)(l), 0, 0);
#else
  *(v8bf*)l = *(const v8bf*)g;
#endif
}

// wait until at most the 6 most recent async ops are still outstanding
__device__ inline void stage_wait_pending() {
#if HAVE_ASYNC
  __builtin_amdgcn_s_wait_asynccnt(6);
#endif
}
__device__ inline void stage_wait_all() {
#if HAVE_ASYNC
  __builtin_amdgcn_s_wait_asynccnt(0);
#endif
}

// ---------------------------------------------------------------------------
// WMMA helpers — contiguous fragment k order, k = 16*half + s (wave32)
// ---------------------------------------------------------------------------
__device__ inline v8f wmma_bf16(v16bf a, v16bf b, v8f c) {
  return __builtin_amdgcn_wmma_f32_16x16x32_bf16(
      /*neg_a=*/false, a, /*neg_b=*/false, b,
      /*c_mod=*/(short)0, c, /*reuse_a=*/false, /*reuse_b=*/false);
}

// A 16x32 bf16 fragment from a row-major [16][ld] tile: per lane two
// contiguous 8-element (16B) runs: K = half*16 + [0,8) and half*16+8 + [0,8).
__device__ inline v16bf load_a_frag(const bf16* base, int ld, int koff) {
  const int lane = threadIdx.x & 31;
  const int half = lane >> 4;
  const int m    = lane & 15;
  const v8bf lo = *(const v8bf*)&base[m * ld + half * 16 + koff];
  const v8bf hi = *(const v8bf*)&base[m * ld + half * 16 + 8 + koff];
  return __builtin_shufflevector(lo, hi, 0, 1, 2, 3, 4, 5, 6, 7, 8, 9, 10, 11,
                                 12, 13, 14, 15);
}

// B 32x16 bf16 fragment from B^T storage: one contiguous 32B load per lane.
__device__ inline v16bf load_bt_frag(const bf16* baseT, int ldk) {
  const int lane = threadIdx.x & 31;
  const int half = lane >> 4;
  const int n    = lane & 15;
  return *(const v16bf*)&baseT[n * ldk + half * 16];
}

// ---------------------------------------------------------------------------
// fp32 -> bf16 conversion (row-major copy)
// ---------------------------------------------------------------------------
__global__ void cvt_kernel(const float* __restrict__ in, bf16* __restrict__ out,
                           long n) {
  long i = (long)blockIdx.x * blockDim.x + threadIdx.x;
  const long stride = (long)gridDim.x * blockDim.x;
  for (; i < n; i += stride) out[i] = (bf16)in[i];
}

// fp32 [K][N] -> bf16 transposed [N][K], 32x32 tiles via LDS, batched over z.
__global__ __launch_bounds__(256) void cvt_t_kernel(const float* __restrict__ in,
                                                    bf16* __restrict__ out,
                                                    int K, int N, long sIn,
                                                    long sOut) {
  __shared__ float tile[32][33];
  const int z = blockIdx.z;
  in += sIn * z;
  out += sOut * z;
  const int kb = blockIdx.y * 32, nb = blockIdx.x * 32;
  const int tx = threadIdx.x & 31, ty = threadIdx.x >> 5;  // 32 x 8
#pragma unroll
  for (int i = 0; i < 32; i += 8)
    tile[ty + i][tx] = in[(long)(kb + ty + i) * N + nb + tx];
  __syncthreads();
#pragma unroll
  for (int i = 0; i < 32; i += 8)
    out[(long)(nb + ty + i) * K + kb + tx] = (bf16)tile[tx][ty + i];
}

// ---------------------------------------------------------------------------
// Tiled bf16 GEMM: C = A[M,K] * B[K,N], B given as B^T [N][K].
// Block tile 128x64, 4 waves (each 32x64 -> 8 WMMAs per K-step of 32).
// LDS double-buffered; tiles staged with async global->LDS (ASYNCcnt).
// Fused bias/ReLU/residual; optional transposed bf16 out (V^T); batched via z.
// ---------------------------------------------------------------------------
__global__ __launch_bounds__(128) void gemm_bf16_kernel(
    const bf16* __restrict__ A, const bf16* __restrict__ BT,
    long strideA, long strideB, long strideC,
    int M, int N, int K,
    const float* __restrict__ bias, const float* __restrict__ res, int relu,
    float* __restrict__ outF, bf16* __restrict__ outB,
    bf16* __restrict__ outBT, long strideCT, int ldt) {
  __shared__ bf16 tA[2][128 * 32];
  __shared__ bf16 tBT[2][64 * 32];

  const int z = blockIdx.z;
  A += strideA * z;
  BT += strideB * z;
  const long cb = strideC * z;

  const int m0 = blockIdx.y * 128;
  const int n0 = blockIdx.x * 64;
  const int tid  = threadIdx.x;
  const int wv   = tid >> 5;
  const int lane = tid & 31;
  const int half = lane >> 4;
  const int ln   = lane & 15;

  // per-thread staging coordinates: 8-element (16B) chunks
  const int r0 = tid >> 2;          // 0..31
  const int k8 = (tid & 3) * 8;     // 0,8,16,24
  const long sK32 = (long)32 * K;   // 32 rows stride

  const bf16* pA = A + (long)(m0 + r0) * K + k8;
  const bf16* pB = BT + (long)(n0 + r0) * K + k8;

#define STAGE_TILE(buf)                                          \
  do {                                                           \
    stage16(pA,            &tA[buf][(r0)*32 + k8]);              \
    stage16(pA + sK32,     &tA[buf][(r0 + 32) * 32 + k8]);       \
    stage16(pA + 2 * sK32, &tA[buf][(r0 + 64) * 32 + k8]);       \
    stage16(pA + 3 * sK32, &tA[buf][(r0 + 96) * 32 + k8]);       \
    stage16(pB,            &tBT[buf][(r0)*32 + k8]);             \
    stage16(pB + sK32,     &tBT[buf][(r0 + 32) * 32 + k8]);      \
    pA += 32;                                                    \
    pB += 32;                                                    \
  } while (0)

  v8f acc[8] = {};

  STAGE_TILE(0);
  const int nk = K >> 5;
  for (int t = 0; t < nk; ++t) {
    const int buf = t & 1;
    if (t + 1 < nk) {
      STAGE_TILE(buf ^ 1);   // 6 async ops for the next tile
      stage_wait_pending();  // tile t's 6 ops complete (in-order)
    } else {
      stage_wait_all();
    }
    __syncthreads();

    const v16bf a0 = load_a_frag(&tA[buf][(wv * 32) * 32], 32, 0);
    const v16bf a1 = load_a_frag(&tA[buf][(wv * 32 + 16) * 32], 32, 0);
    const v16bf b0 = load_bt_frag(&tBT[buf][0], 32);
    const v16bf b1 = load_bt_frag(&tBT[buf][16 * 32], 32);
    const v16bf b2 = load_bt_frag(&tBT[buf][32 * 32], 32);
    const v16bf b3 = load_bt_frag(&tBT[buf][48 * 32], 32);
    acc[0] = wmma_bf16(a0, b0, acc[0]);
    acc[1] = wmma_bf16(a0, b1, acc[1]);
    acc[2] = wmma_bf16(a0, b2, acc[2]);
    acc[3] = wmma_bf16(a0, b3, acc[3]);
    acc[4] = wmma_bf16(a1, b0, acc[4]);
    acc[5] = wmma_bf16(a1, b1, acc[5]);
    acc[6] = wmma_bf16(a1, b2, acc[6]);
    acc[7] = wmma_bf16(a1, b3, acc[7]);
    __syncthreads();
  }
#undef STAGE_TILE

#pragma unroll
  for (int g = 0; g < 2; ++g) {
#pragma unroll
    for (int r = 0; r < 8; ++r) {
      const int gm = m0 + wv * 32 + g * 16 + r + half * 8;
#pragma unroll
      for (int s = 0; s < 4; ++s) {
        const int gn = n0 + s * 16 + ln;
        float xv = acc[g * 4 + s][r];
        if (bias) xv += bias[gn];
        if (relu) xv = fmaxf(xv, 0.0f);
        if (res)  xv += res[(long)gm * N + gn];
        const long idx = cb + (long)gm * N + gn;
        if (outF) outF[idx] = xv;
        if (outB) outB[idx] = (bf16)xv;
        if (outBT) outBT[strideCT * z + (long)gn * ldt + gm] = (bf16)xv;
      }
    }
  }
}

// ---------------------------------------------------------------------------
// Causal flash attention. 1 wave per (head, 16-row q tile); 32-wide key tiles.
// q,k: [H][T][64] bf16.  vT: [H][64][T] bf16.  conc: [T][H*64] bf16.
// ---------------------------------------------------------------------------
__global__ __launch_bounds__(32) void attn_kernel(const bf16* __restrict__ q,
                                                  const bf16* __restrict__ k,
                                                  const bf16* __restrict__ vT,
                                                  bf16* __restrict__ conc) {
  __shared__ bf16 sP[16 * 32];  // P tile staged for A-fragment re-layout

  const int h  = blockIdx.y;
  const int q0 = blockIdx.x * 16;
  const bf16* qh  = q + (long)h * T_ * D_;
  const bf16* kh  = k + (long)h * T_ * D_;
  const bf16* vTh = vT + (long)h * D_ * T_;

  const int lane = threadIdx.x & 31;
  const int half = lane >> 4;
  const int ln   = lane & 15;

  // Q fragments for K-dim chunks d=[0,32) and [32,64)
  const v16bf qa0 = load_a_frag(qh + (long)q0 * D_, D_, 0);
  const v16bf qa1 = load_a_frag(qh + (long)q0 * D_, D_, 32);

  // hoisted per-lane fragment base pointers (advanced by the key-tile stride)
  const bf16* kl = kh + (long)ln * D_ + half * 16;   // +32*D_ per tile
  const bf16* vl = vTh + (long)ln * T_ + half * 16;  // +32 per tile

  float mrow[8], lrow[8];
  v8f o[4] = {};
#pragma unroll
  for (int r = 0; r < 8; ++r) { mrow[r] = -3.0e38f; lrow[r] = 0.0f; }

  constexpr float scale = 0.125f;  // 1/sqrt(64)

  for (int j0 = 0; j0 < q0 + 16; j0 += 32) {
    // ---- S = q . k^T : contiguous 32B B-fragment loads per lane ----
    v8f s0 = {}, s1 = {};
    s0 = wmma_bf16(qa0, *(const v16bf*)(kl), s0);
    s0 = wmma_bf16(qa1, *(const v16bf*)(kl + 32), s0);
    s1 = wmma_bf16(qa0, *(const v16bf*)(kl + 16 * D_), s1);
    s1 = wmma_bf16(qa1, *(const v16bf*)(kl + 16 * D_ + 32), s1);

    __builtin_prefetch(kl + 32 * D_, 0, 1);  // -> global_prefetch_b8
    __builtin_prefetch(vl + 32, 0, 1);

    // ---- causal mask + online softmax in C-fragment layout ----
#pragma unroll
    for (int r = 0; r < 8; ++r) {
      const int rg = q0 + r + half * 8;  // global query row of this register
      float x0 = s0[r] * scale;
      float x1 = s1[r] * scale;
      if (j0 + ln > rg)      x0 = -3.0e38f;
      if (j0 + 16 + ln > rg) x1 = -3.0e38f;
      float mx = fmaxf(x0, x1);
#pragma unroll
      for (int d = 8; d >= 1; d >>= 1) mx = fmaxf(mx, __shfl_xor(mx, d));
      const float mnew  = fmaxf(mrow[r], mx);
      const float alpha = __expf(mrow[r] - mnew);
      const float p0 = __expf(x0 - mnew);
      const float p1 = __expf(x1 - mnew);
      float rs = p0 + p1;
#pragma unroll
      for (int d = 8; d >= 1; d >>= 1) rs += __shfl_xor(rs, d);
      lrow[r] = lrow[r] * alpha + rs;
      mrow[r] = mnew;
#pragma unroll
      for (int s = 0; s < 4; ++s) o[s][r] *= alpha;
      sP[(r + half * 8) * 32 + ln]      = (bf16)p0;
      sP[(r + half * 8) * 32 + 16 + ln] = (bf16)p1;
    }
    __syncthreads();

    // ---- O += P . V : P via LDS as A-frag, V^T gives contiguous B-frags ----
    const v16bf pa = load_a_frag(sP, 32, 0);
#pragma unroll
    for (int s = 0; s < 4; ++s) {
      const v16bf bv = *(const v16bf*)(vl + (long)s * 16 * T_);
      o[s] = wmma_bf16(pa, bv, o[s]);
    }
    __syncthreads();

    kl += 32 * D_;
    vl += 32;
  }

  // ---- normalize and write into concat layout [T, H*64] ----
#pragma unroll
  for (int r = 0; r < 8; ++r) {
    const float inv = 1.0f / lrow[r];
    const int tg = q0 + r + half * 8;
#pragma unroll
    for (int s = 0; s < 4; ++s)
      conc[(long)tg * (H_ * D_) + h * D_ + s * 16 + ln] = (bf16)(o[s][r] * inv);
  }
}

// ---------------------------------------------------------------------------
// Row LayerNorm over E=1024.  One block per row, 256 threads x 4 elements.
// ---------------------------------------------------------------------------
__global__ __launch_bounds__(256) void ln_kernel(const float* __restrict__ in,
                                                 const float* __restrict__ gamma,
                                                 const float* __restrict__ beta,
                                                 float* __restrict__ outF,
                                                 bf16* __restrict__ outB) {
  __shared__ float sh[16];
  const int row = blockIdx.x;
  const int tid = threadIdx.x;
  const float* rp = in + (long)row * E_;

  float vals[4], s = 0.0f, ss = 0.0f;
#pragma unroll
  for (int i = 0; i < 4; ++i) {
    const float x = rp[tid + i * 256];
    vals[i] = x;
    s += x;
    ss += x * x;
  }
#pragma unroll
  for (int d = 16; d >= 1; d >>= 1) {
    s += __shfl_xor(s, d);
    ss += __shfl_xor(ss, d);
  }
  const int w = tid >> 5;
  if ((tid & 31) == 0) { sh[w] = s; sh[8 + w] = ss; }
  __syncthreads();
  if (tid == 0) {
    float S = 0.0f, SS = 0.0f;
    for (int i = 0; i < 8; ++i) { S += sh[i]; SS += sh[8 + i]; }
    sh[0] = S;
    sh[8] = SS;
  }
  __syncthreads();
  const float mean = sh[0] * (1.0f / E_);
  const float var  = sh[8] * (1.0f / E_) - mean * mean;
  const float inv  = rsqrtf(var + 1e-5f) * gamma[0];
#pragma unroll
  for (int i = 0; i < 4; ++i) {
    const int c = tid + i * 256;
    const float y = (vals[i] - mean) * inv + beta[c];
    const long idx = (long)row * E_ + c;
    if (outF) outF[idx] = y;
    if (outB) outB[idx] = (bf16)y;
  }
}

// ---------------------------------------------------------------------------
// Host-side orchestration
// ---------------------------------------------------------------------------
extern "C" void kernel_launch(void* const* d_in, const int* in_sizes, int n_in,
                              void* d_out, int out_size, void* d_ws,
                              size_t ws_size, hipStream_t stream) {
  (void)in_sizes; (void)n_in; (void)out_size; (void)ws_size;

  const float* x      = (const float*)d_in[0];
  const float* wq     = (const float*)d_in[1];
  const float* wk     = (const float*)d_in[2];
  const float* wv     = (const float*)d_in[3];
  const float* pool_w = (const float*)d_in[4];
  const float* l1_w   = (const float*)d_in[5];
  const float* l1_b   = (const float*)d_in[6];
  const float* l2_w   = (const float*)d_in[7];
  const float* l2_b   = (const float*)d_in[8];
  const float* gamma  = (const float*)d_in[9];
  const float* beta   = (const float*)d_in[10];
  float* out = (float*)d_out;

  size_t off = 0;
  auto alloc = [&](size_t bytes) -> char* {
    off = (off + 255) & ~(size_t)255;
    char* p = (char*)d_ws + off;
    off += bytes;
    return p;
  };
  bf16* xb    = (bf16*)alloc((size_t)T_ * E_ * 2);
  bf16* wqT   = (bf16*)alloc((size_t)H_ * D_ * E_ * 2);   // [H][D][E]
  bf16* wkT   = (bf16*)alloc((size_t)H_ * D_ * E_ * 2);
  bf16* wvT   = (bf16*)alloc((size_t)H_ * D_ * E_ * 2);
  bf16* poolT = (bf16*)alloc((size_t)E_ * H_ * D_ * 2);   // [E][HD]
  bf16* l1T   = (bf16*)alloc((size_t)F_ * E_ * 2);        // [F][E]
  bf16* l2T   = (bf16*)alloc((size_t)E_ * F_ * 2);        // [E][F]
  bf16* qb    = (bf16*)alloc((size_t)H_ * T_ * D_ * 2);   // [H][T][D]
  bf16* kb    = (bf16*)alloc((size_t)H_ * T_ * D_ * 2);   // [H][T][D]
  bf16* vT    = (bf16*)alloc((size_t)H_ * D_ * T_ * 2);   // [H][D][T]
  bf16* conc  = (bf16*)alloc((size_t)T_ * H_ * D_ * 2);   // [T][HD]
  float* tmpF = (float*)alloc((size_t)T_ * E_ * 4);
  float* hbuf = (float*)alloc((size_t)T_ * E_ * 4);
  bf16*  hb   = (bf16*)alloc((size_t)T_ * E_ * 2);
  bf16*  act  = (bf16*)alloc((size_t)T_ * F_ * 2);

  auto cvt = [&](const float* src, bf16* dst, long n) {
    const int grid = (int)((n + 1023) / 1024);
    cvt_kernel<<<grid, 256, 0, stream>>>(src, dst, n);
  };
  auto cvtT = [&](const float* src, bf16* dst, int K, int N, int batch) {
    dim3 g(N / 32, K / 32, batch);
    cvt_t_kernel<<<g, 256, 0, stream>>>(src, dst, K, N, (long)K * N,
                                        (long)N * K);
  };
  cvt(x, xb, (long)T_ * E_);
  cvtT(wq, wqT, E_, D_, H_);
  cvtT(wk, wkT, E_, D_, H_);
  cvtT(wv, wvT, E_, D_, H_);
  cvtT(pool_w, poolT, H_ * D_, E_, 1);
  cvtT(l1_w, l1T, E_, F_, 1);
  cvtT(l2_w, l2T, F_, E_, 1);

  // QKV projections, batched over heads: [T,E] x [E,64] per head
  {
    dim3 g(D_ / 64, T_ / 128, H_);
    gemm_bf16_kernel<<<g, 128, 0, stream>>>(
        xb, wqT, 0, (long)D_ * E_, (long)T_ * D_, T_, D_, E_, nullptr, nullptr,
        0, nullptr, qb, nullptr, 0, 0);
    gemm_bf16_kernel<<<g, 128, 0, stream>>>(
        xb, wkT, 0, (long)D_ * E_, (long)T_ * D_, T_, D_, E_, nullptr, nullptr,
        0, nullptr, kb, nullptr, 0, 0);
    gemm_bf16_kernel<<<g, 128, 0, stream>>>(
        xb, wvT, 0, (long)D_ * E_, 0, T_, D_, E_, nullptr, nullptr, 0, nullptr,
        nullptr, vT, (long)D_ * T_, T_);
  }

  // causal flash attention -> conc [T, H*D]
  attn_kernel<<<dim3(T_ / 16, H_), 32, 0, stream>>>(qb, kb, vT, conc);

  // pool: conc @ pool_w + x -> tmpF;  LN -> hbuf (f32) + hb (bf16)
  gemm_bf16_kernel<<<dim3(E_ / 64, T_ / 128, 1), 128, 0, stream>>>(
      conc, poolT, 0, 0, 0, T_, E_, H_ * D_, nullptr, x, 0, tmpF, nullptr,
      nullptr, 0, 0);
  ln_kernel<<<T_, 256, 0, stream>>>(tmpF, gamma, beta, hbuf, hb);

  // FFN: relu(h @ l1 + b1) @ l2 + b2 + h -> tmpF;  LN -> out
  gemm_bf16_kernel<<<dim3(F_ / 64, T_ / 128, 1), 128, 0, stream>>>(
      hb, l1T, 0, 0, 0, T_, F_, E_, l1_b, nullptr, 1, nullptr, act, nullptr, 0,
      0);
  gemm_bf16_kernel<<<dim3(E_ / 64, T_ / 128, 1), 128, 0, stream>>>(
      act, l2T, 0, 0, 0, T_, E_, F_, l2_b, hbuf, 0, tmpF, nullptr, nullptr, 0,
      0);
  ln_kernel<<<T_, 256, 0, stream>>>(tmpF, gamma, beta, out, nullptr);
}
